// Block_37237366456639
// MI455X (gfx1250) — matrix operand order repaired and checked by reading, not compiled, not run on hardware
//
#include <hip/hip_runtime.h>
#include <hip/hip_bf16.h>

typedef __attribute__((ext_vector_type(16))) _Float16 v16h;
typedef __attribute__((ext_vector_type(8)))  float    v8f;
typedef __attribute__((ext_vector_type(4)))  unsigned int u32x4;
typedef __attribute__((ext_vector_type(8)))  int      i32x8;
typedef __attribute__((ext_vector_type(4)))  int      i32x4;

union F16x16 { v16h v; uint4 q[2]; };

__device__ __forceinline__ v8f wmma_f16(const F16x16& a, const F16x16& b, v8f c) {
    return __builtin_amdgcn_wmma_f32_16x16x32_f16(
        false, a.v, false, b.v, (short)0, c, false, false);
}

#if defined(__AMDGCN__) && __has_builtin(__builtin_amdgcn_tensor_load_to_lds)
#define USE_TDM 1
#else
#define USE_TDM 0
#endif

#if USE_TDM
// Issue one 2D TDM tile load: tile_dim0 = 32 halfs (contiguous), tile_dim1 =
// 128 rows, row stride = K elements, LDS rows padded 32+8 halfs (pad 4 dwords
// after every 16 dwords). Descriptor packing per CDNA5 ISA ch.8.3/8.4.
__device__ __forceinline__ void tdm_load_tile(const _Float16* gptr, unsigned lds_addr,
                                              int K, int dim1) {
    unsigned long long ga = (unsigned long long)(size_t)gptr;
    u32x4 g0;
    g0.x = 1u;                                    // count=1, user mode
    g0.y = lds_addr;                              // LDS byte address
    g0.z = (unsigned)ga;                          // global addr [31:0]
    g0.w = (unsigned)((ga >> 32) & 0x1FFFFFFu) | (2u << 30);  // [56:32] + type=2
    const unsigned dw0 = (1u << 16)   // data_size = 2 bytes
                       | (1u << 20)   // pad_enable
                       | (3u << 22)   // pad_interval: 16 dwords (32 halfs)
                       | (3u << 25);  // pad_amount:   4 dwords  (8 halfs)
    const unsigned uK = (unsigned)K, uD1 = (unsigned)dim1;
    const unsigned dw1 = (uK & 0xFFFFu) << 16;                         // tensor_dim0 lo
    const unsigned dw2 = ((uK >> 16) & 0xFFFFu) | ((uD1 & 0xFFFFu) << 16); // dim0 hi, dim1 lo
    const unsigned dw3 = ((uD1 >> 16) & 0xFFFFu) | (32u << 16);        // dim1 hi, tile_dim0=32
    const unsigned dw4 = 128u;                                         // tile_dim1=128
    const unsigned dw5 = uK;                                           // dim0 stride lo
    i32x8 g1 = {(int)dw0, (int)dw1, (int)dw2, (int)dw3,
                (int)dw4, (int)dw5, 0, 0};
    i32x4 z4 = {0, 0, 0, 0};
#if __clang_major__ >= 23
    i32x8 z8 = {0, 0, 0, 0, 0, 0, 0, 0};
    __builtin_amdgcn_tensor_load_to_lds(g0, g1, z4, z4, z8, 0);
#else
    __builtin_amdgcn_tensor_load_to_lds(g0, g1, z4, z4, 0);
#endif
}
#endif

// ---------------------------------------------------------------------------
// f32 -> f16 weight conversion WITH transpose: out[n*K + k] = in[k*N + n].
// Makes both GEMM operands row-contiguous in K so TDM can stage them.
// ---------------------------------------------------------------------------
__global__ __launch_bounds__(256) void convert_transpose_f16(
    const float* __restrict__ in, _Float16* __restrict__ out, int K, int N) {
    size_t total = (size_t)K * N;
    size_t i = (size_t)blockIdx.x * blockDim.x + threadIdx.x;
    size_t stride = (size_t)gridDim.x * blockDim.x;
    for (; i < total; i += stride) {
        size_t k = i / N, n = i - k * N;
        out[n * (size_t)K + k] = (_Float16)in[i];
    }
}

// ---------------------------------------------------------------------------
// LayerNorm over DIM=1024, emits f16 activations (one block per token)
// ---------------------------------------------------------------------------
__global__ __launch_bounds__(256) void layernorm_to_f16(
    const float* __restrict__ x, const float* __restrict__ gw,
    const float* __restrict__ bw, _Float16* __restrict__ out) {
    __shared__ float red[8];
    const int t = blockIdx.x;
    const float* xr = x + (size_t)t * 1024;
    float v[4];
    float s = 0.f;
#pragma unroll
    for (int i = 0; i < 4; ++i) { v[i] = xr[threadIdx.x + i * 256]; s += v[i]; }
#pragma unroll
    for (int o = 16; o; o >>= 1) s += __shfl_xor(s, o, 32);
    if ((threadIdx.x & 31) == 0) red[threadIdx.x >> 5] = s;
    __syncthreads();
    s = 0.f;
#pragma unroll
    for (int i = 0; i < 8; ++i) s += red[i];
    const float mu = s * (1.0f / 1024.0f);
    __syncthreads();
    float q = 0.f;
#pragma unroll
    for (int i = 0; i < 4; ++i) { float d = v[i] - mu; q += d * d; }
#pragma unroll
    for (int o = 16; o; o >>= 1) q += __shfl_xor(q, o, 32);
    if ((threadIdx.x & 31) == 0) red[threadIdx.x >> 5] = q;
    __syncthreads();
    q = 0.f;
#pragma unroll
    for (int i = 0; i < 8; ++i) q += red[i];
    const float inv = rsqrtf(q * (1.0f / 1024.0f) + 1e-5f);
#pragma unroll
    for (int i = 0; i < 4; ++i) {
        int c = threadIdx.x + i * 256;
        out[(size_t)t * 1024 + c] = (_Float16)((v[i] - mu) * inv * gw[c] + bw[c]);
    }
}

// ---------------------------------------------------------------------------
// Tiled WMMA GEMM with TDM double-buffered LDS staging.
//   C[M,N] = A[M,K](f16) @ Bt[N,K](f16, pre-transposed) + bias, fused epilogue
// Block tile 128x128, K-step 32, 256 threads (8 waves), wave tile 64x32.
// MODE 0: out f16 = acc + bias              (QKV)
// MODE 1: out f32 = acc + bias + resid      (proj / FF2 + residual)
// MODE 2: out f16 = gelu(acc + bias)        (FF1)
// ---------------------------------------------------------------------------
#define LDST 40   // LDS row stride in halfs (32 + 8 pad)

template <int MODE>
__global__ __launch_bounds__(256, 2) void gemm_wmma(
    const _Float16* __restrict__ A, const _Float16* __restrict__ Bt,
    const float* __restrict__ bias, const float* __restrict__ resid,
    float* __restrict__ outF, _Float16* __restrict__ outH,
    int M, int N, int K) {
    __shared__ __align__(16) _Float16 As[2][128 * LDST];
    __shared__ __align__(16) _Float16 Bs[2][128 * LDST];

    const int tid  = threadIdx.x;
    const int lane = tid & 31;
    const int wave = tid >> 5;
    const int half = lane >> 4, ln = lane & 15;
    const int waveM = wave & 1, waveN = wave >> 1;
    const int bm = blockIdx.y * 128, bn = blockIdx.x * 128;

    v8f acc[4][2] = {};

    // stage one K-tile pair (A rows bm.., Bt rows bn..) into buffer `buf`
    auto stage = [&](int buf, int k0) {
#if USE_TDM
        if (tid < 32) {   // wave 0 issues; TENSORcnt tracks completion
            tdm_load_tile(A  + (size_t)bm * K + k0,
                          (unsigned)(size_t)(void*)As[buf], K, M);
            tdm_load_tile(Bt + (size_t)bn * K + k0,
                          (unsigned)(size_t)(void*)Bs[buf], K, N);
        }
#else
#pragma unroll
        for (int i = 0; i < 2; ++i) {
            int ch = tid + i * 256;
            int row = ch >> 2, c = (ch & 3) * 8;
            *(uint4*)(&As[buf][row * LDST + c]) =
                *(const uint4*)(A + (size_t)(bm + row) * K + k0 + c);
            *(uint4*)(&Bs[buf][row * LDST + c]) =
                *(const uint4*)(Bt + (size_t)(bn + row) * K + k0 + c);
        }
#endif
    };

    stage(0, 0);
    int buf = 0;
    for (int k0 = 0; k0 < K; k0 += 32, buf ^= 1) {
        const bool more = (k0 + 32 < K);
        if (more) stage(buf ^ 1, k0 + 32);
#if USE_TDM
        if (tid < 32) {   // immediate operand required: constant in each arm
            if (more) __builtin_amdgcn_s_wait_tensorcnt(2);
            else      __builtin_amdgcn_s_wait_tensorcnt(0);
        }
#endif
        __syncthreads();

        // fragments per ISA layouts (A: K = half*8 / 16+half*8; B: K = half*8 / 16+half*8)
        F16x16 aF[4], bF[2];
#pragma unroll
        for (int mt = 0; mt < 4; ++mt) {
            const _Float16* r = &As[buf][(waveM * 64 + mt * 16 + ln) * LDST];
            aF[mt].q[0] = *(const uint4*)(r + half * 8);
            aF[mt].q[1] = *(const uint4*)(r + 16 + half * 8);
        }
#pragma unroll
        for (int nt = 0; nt < 2; ++nt) {
            const _Float16* r = &Bs[buf][(waveN * 32 + nt * 16 + ln) * LDST];
            bF[nt].q[0] = *(const uint4*)(r + half * 8);
            bF[nt].q[1] = *(const uint4*)(r + 16 + half * 8);
        }
#pragma unroll
        for (int mt = 0; mt < 4; ++mt)
#pragma unroll
            for (int nt = 0; nt < 2; ++nt)
                acc[mt][nt] = wmma_f16(aF[mt], bF[nt], acc[mt][nt]);

        __syncthreads();   // all reads done before this buffer is re-staged
    }

    // epilogue: C layout -> row = r + half*8, col = lane%16
#pragma unroll
    for (int mt = 0; mt < 4; ++mt) {
#pragma unroll
        for (int nt = 0; nt < 2; ++nt) {
            const int col = bn + waveN * 32 + nt * 16 + ln;
            const float bv = bias[col];
#pragma unroll
            for (int r = 0; r < 8; ++r) {
                const int rowg = bm + waveM * 64 + mt * 16 + half * 8 + r;
                const size_t idx = (size_t)rowg * N + col;
                float val = acc[mt][nt][r] + bv;
                if constexpr (MODE == 0) {
                    outH[idx] = (_Float16)val;
                } else if constexpr (MODE == 1) {
                    outF[idx] = val + resid[idx];
                } else {
                    outH[idx] = (_Float16)(0.5f * val * (1.0f + erff(val * 0.70710678f)));
                }
            }
        }
    }
}

// ---------------------------------------------------------------------------
// Flash attention, causal, head_dim=64, T=2048. 4 waves/block, each wave owns
// a 16-query tile; 32-key blocks staged in LDS; online softmax; all matmuls
// via v_wmma_f32_16x16x32_f16.  grid = (T/64, NH, B), block = 128
// ---------------------------------------------------------------------------
__global__ __launch_bounds__(128) void flash_attn(const _Float16* __restrict__ qkv,
                                                  _Float16* __restrict__ out) {
    __shared__ __align__(16) _Float16 Ks[32 * 72];       // K rows as-is
    __shared__ __align__(16) _Float16 Vt[64 * 40];       // V transposed [d][k]
    __shared__ __align__(16) _Float16 Pb[4 * 16 * 40];   // per-wave P reshape

    const int tid  = threadIdx.x;
    const int lane = tid & 31, wave = tid >> 5;
    const int half = lane >> 4, ln = lane & 15;
    const int qBase = blockIdx.x * 64;
    const int h = blockIdx.y;
    const size_t bt0 = (size_t)blockIdx.z * 2048;
    const int waveQ = qBase + wave * 16;

    F16x16 qF[2];
    {
        const _Float16* qr = qkv + (bt0 + waveQ + ln) * 3072 + h * 64;
        qF[0].q[0] = *(const uint4*)(qr + half * 8);
        qF[0].q[1] = *(const uint4*)(qr + 16 + half * 8);
        qF[1].q[0] = *(const uint4*)(qr + 32 + half * 8);
        qF[1].q[1] = *(const uint4*)(qr + 48 + half * 8);
    }

    v8f accO[4] = {};
    float mrow[8], lrow[8];
#pragma unroll
    for (int r = 0; r < 8; ++r) { mrow[r] = -1e30f; lrow[r] = 0.f; }

    const int kEnd = qBase + 64;
    for (int kb = 0; kb < kEnd; kb += 32) {
        __syncthreads();
#pragma unroll
        for (int i = 0; i < 2; ++i) {
            int ch  = tid + i * 128;
            int key = ch >> 3, c = (ch & 7) * 8;
            const size_t rowb = (bt0 + kb + key) * 3072 + h * 64 + c;
            uint4 dk = *(const uint4*)(qkv + rowb + 1024);
            *(uint4*)(&Ks[key * 72 + c]) = dk;
            uint4 dv = *(const uint4*)(qkv + rowb + 2048);
            _Float16 tmp[8];
            *(uint4*)tmp = dv;
#pragma unroll
            for (int j = 0; j < 8; ++j) Vt[(c + j) * 40 + key] = tmp[j];
        }
        __syncthreads();

        if (kb <= waveQ + 15) {
            v8f S[2] = {};
#pragma unroll
            for (int nt = 0; nt < 2; ++nt) {
                const _Float16* kr = &Ks[(nt * 16 + ln) * 72];
                F16x16 k0, k1;
                k0.q[0] = *(const uint4*)(kr + half * 8);
                k0.q[1] = *(const uint4*)(kr + 16 + half * 8);
                k1.q[0] = *(const uint4*)(kr + 32 + half * 8);
                k1.q[1] = *(const uint4*)(kr + 48 + half * 8);
                S[nt] = wmma_f16(qF[0], k0, S[nt]);
                S[nt] = wmma_f16(qF[1], k1, S[nt]);
            }
            float mnew[8], sc[8], psum[8];
#pragma unroll
            for (int r = 0; r < 8; ++r) {
                const int q = waveQ + half * 8 + r;
                float s0 = S[0][r] * 0.125f;
                float s1 = S[1][r] * 0.125f;
                if (kb + ln > q)      s0 = -1e30f;
                if (kb + 16 + ln > q) s1 = -1e30f;
                S[0][r] = s0; S[1][r] = s1;
                float mx = fmaxf(s0, s1);
#pragma unroll
                for (int o = 1; o < 16; o <<= 1) mx = fmaxf(mx, __shfl_xor(mx, o, 32));
                mnew[r] = fmaxf(mrow[r], mx);
                sc[r]   = __expf(mrow[r] - mnew[r]);
                mrow[r] = mnew[r];
            }
#pragma unroll
            for (int r = 0; r < 8; ++r) {
                float p0 = __expf(S[0][r] - mnew[r]);
                float p1 = __expf(S[1][r] - mnew[r]);
                _Float16* pw = &Pb[wave * 640 + (half * 8 + r) * 40];
                pw[ln]      = (_Float16)p0;
                pw[16 + ln] = (_Float16)p1;
                float ps = p0 + p1;
#pragma unroll
                for (int o = 1; o < 16; o <<= 1) ps += __shfl_xor(ps, o, 32);
                psum[r] = ps;
            }
#pragma unroll
            for (int r = 0; r < 8; ++r) lrow[r] = lrow[r] * sc[r] + psum[r];
#pragma unroll
            for (int nt = 0; nt < 4; ++nt)
#pragma unroll
                for (int r = 0; r < 8; ++r) accO[nt][r] *= sc[r];

            asm volatile("s_wait_dscnt 0" ::: "memory");  // same-wave P RAW

            F16x16 pF;
            {
                const _Float16* pr = &Pb[wave * 640 + ln * 40];
                pF.q[0] = *(const uint4*)(pr + half * 8);
                pF.q[1] = *(const uint4*)(pr + 16 + half * 8);
            }
#pragma unroll
            for (int nt = 0; nt < 4; ++nt) {
                const _Float16* vr = &Vt[(nt * 16 + ln) * 40];
                F16x16 vF;
                vF.q[0] = *(const uint4*)(vr + half * 8);
                vF.q[1] = *(const uint4*)(vr + 16 + half * 8);
                accO[nt] = wmma_f16(pF, vF, accO[nt]);
            }
        }
    }

#pragma unroll
    for (int r = 0; r < 8; ++r) {
        const float inv = 1.0f / lrow[r];
        const size_t row = (bt0 + waveQ + half * 8 + r) * 1024 + h * 64;
#pragma unroll
        for (int nt = 0; nt < 4; ++nt)
            out[row + nt * 16 + ln] = (_Float16)(accO[nt][r] * inv);
    }
}

// ---------------------------------------------------------------------------
// Orchestration
// ---------------------------------------------------------------------------
extern "C" void kernel_launch(void* const* d_in, const int* in_sizes, int n_in,
                              void* d_out, int out_size, void* d_ws, size_t ws_size,
                              hipStream_t stream) {
    (void)in_sizes; (void)n_in; (void)out_size; (void)ws_size;
    const int DIM = 1024, T = 2048, Bb = 4, NH = 16, HID = 4096;
    const size_t TOK = (size_t)Bb * T;  // 8192

    const float* x      = (const float*)d_in[0];
    const float* ln1_g  = (const float*)d_in[1];
    const float* ln1_b  = (const float*)d_in[2];
    const float* qkv_w  = (const float*)d_in[3];
    const float* qkv_b  = (const float*)d_in[4];
    const float* proj_w = (const float*)d_in[5];
    const float* proj_b = (const float*)d_in[6];
    const float* ln2_g  = (const float*)d_in[7];
    const float* ln2_b  = (const float*)d_in[8];
    const float* ff1_w  = (const float*)d_in[9];
    const float* ff1_b  = (const float*)d_in[10];
    const float* ff2_w  = (const float*)d_in[11];
    const float* ff2_b  = (const float*)d_in[12];
    float* out = (float*)d_out;

    char* p = (char*)d_ws;
    auto carve = [&](size_t bytes) -> char* {
        char* r = p;
        p += (bytes + 255) & ~(size_t)255;
        return r;
    };
    // transposed f16 weights: Wt[N][K]
    _Float16* qkvWt = (_Float16*)carve((size_t)DIM * 3 * DIM * 2);
    _Float16* projWt = (_Float16*)carve((size_t)DIM * DIM * 2);
    _Float16* ff1Wt = (_Float16*)carve((size_t)DIM * HID * 2);
    _Float16* ff2Wt = (_Float16*)carve((size_t)HID * DIM * 2);
    _Float16* xn1   = (_Float16*)carve(TOK * DIM * 2);
    _Float16* qkvO  = (_Float16*)carve(TOK * 3 * DIM * 2);
    _Float16* attnO = (_Float16*)carve(TOK * DIM * 2);
    float*    x1    = (float*)   carve(TOK * DIM * 4);
    _Float16* xn2   = (_Float16*)carve(TOK * DIM * 2);
    _Float16* h1    = (_Float16*)carve(TOK * HID * 2);

    // weight conversion + transpose: [K,N] f32 -> [N,K] f16
    convert_transpose_f16<<<4096, 256, 0, stream>>>(qkv_w,  qkvWt, DIM, 3 * DIM);
    convert_transpose_f16<<<4096, 256, 0, stream>>>(proj_w, projWt, DIM, DIM);
    convert_transpose_f16<<<4096, 256, 0, stream>>>(ff1_w,  ff1Wt, DIM, HID);
    convert_transpose_f16<<<4096, 256, 0, stream>>>(ff2_w,  ff2Wt, HID, DIM);

    // LN1
    layernorm_to_f16<<<(int)TOK, 256, 0, stream>>>(x, ln1_g, ln1_b, xn1);
    // QKV: [8192,1024] @ [1024,3072] -> f16
    gemm_wmma<0><<<dim3(3 * DIM / 128, TOK / 128), 256, 0, stream>>>(
        xn1, qkvWt, qkv_b, nullptr, nullptr, qkvO, (int)TOK, 3 * DIM, DIM);
    // attention
    flash_attn<<<dim3(T / 64, NH, Bb), 128, 0, stream>>>(qkvO, attnO);
    // proj + residual(x) -> x1 (f32)
    gemm_wmma<1><<<dim3(DIM / 128, TOK / 128), 256, 0, stream>>>(
        attnO, projWt, proj_b, x, x1, nullptr, (int)TOK, DIM, DIM);
    // LN2
    layernorm_to_f16<<<(int)TOK, 256, 0, stream>>>(x1, ln2_g, ln2_b, xn2);
    // FF1 + GELU -> f16
    gemm_wmma<2><<<dim3(HID / 128, TOK / 128), 256, 0, stream>>>(
        xn2, ff1Wt, ff1_b, nullptr, nullptr, h1, (int)TOK, HID, DIM);
    // FF2 + residual(x1) -> out (f32)
    gemm_wmma<1><<<dim3(DIM / 128, TOK / 128), 256, 0, stream>>>(
        h1, ff2Wt, ff2_b, x1, out, nullptr, (int)TOK, DIM, HID);
}
